// MoE_layer_flux_53601191854777
// MI455X (gfx1250) — matrix operand order repaired and checked by reading, not compiled, not run on hardware
//
#include <hip/hip_runtime.h>

// ---------------- problem constants (match reference) ----------------
#define NTOK   8192
#define HIDDEN 2048
#define FFN    2048
#define NEXP   8
#define TOPK   2
#define MROWS  (NTOK * TOPK)          // 16384 scattered rows
#define CAP    (MROWS / NEXP)         // 2048 rows per expert

// ---------------- vector types ----------------
typedef __attribute__((ext_vector_type(8)))  unsigned short v8us;
typedef __attribute__((ext_vector_type(16))) unsigned short v16us;
typedef __attribute__((ext_vector_type(16))) __bf16         v16bf;
typedef __attribute__((ext_vector_type(8)))  float          v8f;

// float -> bf16 bits, round-to-nearest-even
static __device__ __forceinline__ unsigned short f2bfu(float f) {
    unsigned u = __builtin_bit_cast(unsigned, f);
    u += 0x7FFFu + ((u >> 16) & 1u);
    return (unsigned short)(u >> 16);
}

static __device__ __forceinline__ float gelu_exact(float x) {
    return 0.5f * x * (1.0f + erff(x * 0.70710678118654752f));
}

// ---------------------------------------------------------------------
// Kernel 0: one-time fp32 -> bf16 weight downconvert (keeps the GEMM
// inner loop free of conversion VALU). 16 elems / thread, vectorized.
// ---------------------------------------------------------------------
__global__ void f32_to_bf16_kernel(const float* __restrict__ src,
                                   unsigned short* __restrict__ dst,
                                   long long n) {
    long long i = ((long long)blockIdx.x * blockDim.x + threadIdx.x) * 16;
    if (i + 16 > n) return;
    const float4* s = (const float4*)(src + i);
    float4 a = s[0], b = s[1], c = s[2], d = s[3];
    v8us lo, hi;
    lo[0] = f2bfu(a.x); lo[1] = f2bfu(a.y); lo[2] = f2bfu(a.z); lo[3] = f2bfu(a.w);
    lo[4] = f2bfu(b.x); lo[5] = f2bfu(b.y); lo[6] = f2bfu(b.z); lo[7] = f2bfu(b.w);
    hi[0] = f2bfu(c.x); hi[1] = f2bfu(c.y); hi[2] = f2bfu(c.z); hi[3] = f2bfu(c.w);
    hi[4] = f2bfu(d.x); hi[5] = f2bfu(d.y); hi[6] = f2bfu(d.z); hi[7] = f2bfu(d.w);
    *(v8us*)(dst + i)     = lo;
    *(v8us*)(dst + i + 8) = hi;
}

// ---------------------------------------------------------------------
// Kernel 1: scatter-dispatch + fp32->bf16 downconvert + inverse map.
// One block per scattered row (t,k). 256 threads x 8 elems = 2048 = HIDDEN.
// ---------------------------------------------------------------------
__global__ void moe_scatter_bf16(const float* __restrict__ x,
                                 const int*   __restrict__ sidx,
                                 unsigned short* __restrict__ xs,
                                 int* __restrict__ inv) {
    const int row = blockIdx.x;               // 0..MROWS-1 over (t,k)
    const int t   = row / TOPK;
    const int dst = sidx[row];
    const float* __restrict__ src = x + (size_t)t * HIDDEN;
    unsigned short* __restrict__ drow = xs + (size_t)dst * HIDDEN;

    const int c = threadIdx.x * 8;
    v8us o;
#pragma unroll
    for (int i = 0; i < 8; ++i) o[i] = f2bfu(src[c + i]);
    *(v8us*)&drow[c] = o;

    if (threadIdx.x == 0) inv[dst] = t;
}

// ---------------------------------------------------------------------
// Tiled WMMA GEMM with double-buffered LDS. C[128x128] per 256-thread
// block (8 wave32 waves). Wave grid 2(M) x 4(N); each wave owns 64x32 =
// 4x2 tiles of v_wmma_f32_16x16x32_bf16. Both A and B are bf16 streams.
// FC1 epilogue: exact GELU -> bf16 h. FC2 epilogue: gather-RS combine
// via atomicAdd into out[inv[row]] (exactly TOPK=2 adds onto zero ->
// bit-deterministic by fp-add commutativity).
// ---------------------------------------------------------------------
#define LDT 40   // padded LDS row stride (elements): 80B, 16B-aligned, bank-skewed

template <bool FC1>
__global__ void moe_wmma_gemm(const unsigned short* __restrict__ Abuf,
                              const unsigned short* __restrict__ Wb16,
                              unsigned short* __restrict__ Hout,
                              const int* __restrict__ inv,
                              float* __restrict__ Out) {
    __shared__ unsigned short As[2][128 * LDT];
    __shared__ unsigned short Bs[2][128 * LDT];

    const int tid  = threadIdx.x;
    const int wave = tid >> 5;
    const int lane = tid & 31;
    const int wm   = wave & 1;       // 2 waves along M
    const int wn   = wave >> 1;      // 4 waves along N

    const int mt = blockIdx.x;       // 16 M-tiles of 128 (CAP=2048)
    const int nt = blockIdx.y;       // 16 N-tiles of 128 (2048 out features)
    const int e  = blockIdx.z;       // expert

    const unsigned short* __restrict__ Ab = Abuf + (size_t)e * CAP * 2048
                                                 + (size_t)(mt * 128) * 2048;
    const unsigned short* __restrict__ Wb = Wb16 + (size_t)e * 2048 * 2048
                                                 + (size_t)(nt * 128) * 2048;

    v8f acc[4][2] = {};

    // per-thread tile-staging coords: 2 threads per row, 16 bf16 per thread
    const int lr = tid >> 1;
    const int lc = (tid & 1) * 16;
    const size_t srow = (size_t)lr * 2048 + lc;

    // fragment coords (per CDNA5 ISA layouts)
    const int am  = lane & 15;
    const int kb  = (lane >> 4) << 3;   // A: K chunks kb..kb+7 and kb+16..kb+23
    const int bn  = lane & 15;
    const int bkb = (lane >> 4) << 3;   // B: K chunks bkb..bkb+7 and bkb+16..bkb+23

    // ---- prologue: stage tile k0=0 into buffer 0 ----
    {
        const v8us* sa = (const v8us*)(Ab + srow);
        const v8us* sb = (const v8us*)(Wb + srow);
        *(v8us*)&As[0][lr * LDT + lc]     = sa[0];
        *(v8us*)&As[0][lr * LDT + lc + 8] = sa[1];
        *(v8us*)&Bs[0][lr * LDT + lc]     = sb[0];
        *(v8us*)&Bs[0][lr * LDT + lc + 8] = sb[1];
    }
    __syncthreads();

    int cur = 0;
    for (int k0 = 0; k0 < 2048; k0 += 32) {
        const bool hasNext = (k0 + 32) < 2048;

        // ---- issue next tile's global loads early (hide under WMMA) ----
        v8us ra0, ra1, rb0, rb1;
        if (hasNext) {
            const v8us* sa = (const v8us*)(Ab + srow + k0 + 32);
            const v8us* sb = (const v8us*)(Wb + srow + k0 + 32);
            ra0 = sa[0]; ra1 = sa[1];
            rb0 = sb[0]; rb1 = sb[1];
            if (k0 + 64 < 2048) {  // warm L2 for tile after next
                __builtin_prefetch(Ab + srow + k0 + 64, 0, 0);
                __builtin_prefetch(Wb + srow + k0 + 64, 0, 0);
            }
        }

        // ---- fragments from current buffer ----
        v16bf afr[4], bfr[2];
#pragma unroll
        for (int i = 0; i < 4; ++i) {
            const unsigned short* p = &As[cur][(wm * 64 + i * 16 + am) * LDT + kb];
            v8us lo = *(const v8us*)p;
            v8us hi = *(const v8us*)(p + 16);
            v16us w16 = __builtin_shufflevector(lo, hi, 0,1,2,3,4,5,6,7,
                                                        8,9,10,11,12,13,14,15);
            afr[i] = __builtin_bit_cast(v16bf, w16);
        }
#pragma unroll
        for (int j = 0; j < 2; ++j) {
            const unsigned short* p = &Bs[cur][(wn * 32 + j * 16 + bn) * LDT + bkb];
            v8us lo = *(const v8us*)p;
            v8us hi = *(const v8us*)(p + 16);
            v16us w16 = __builtin_shufflevector(lo, hi, 0,1,2,3,4,5,6,7,
                                                        8,9,10,11,12,13,14,15);
            bfr[j] = __builtin_bit_cast(v16bf, w16);
        }

#pragma unroll
        for (int i = 0; i < 4; ++i)
#pragma unroll
            for (int j = 0; j < 2; ++j)
                acc[i][j] = __builtin_amdgcn_wmma_f32_16x16x32_bf16(
                    /*neg_a=*/false, afr[i], /*neg_b=*/false, bfr[j],
                    /*c_mod=*/(short)0, acc[i][j],
                    /*reuse_a=*/false, /*reuse_b=*/false);

        // ---- drain staged regs into the other buffer ----
        if (hasNext) {
            const int nxt = cur ^ 1;
            *(v8us*)&As[nxt][lr * LDT + lc]     = ra0;
            *(v8us*)&As[nxt][lr * LDT + lc + 8] = ra1;
            *(v8us*)&Bs[nxt][lr * LDT + lc]     = rb0;
            *(v8us*)&Bs[nxt][lr * LDT + lc + 8] = rb1;
        }
        __syncthreads();
        cur ^= 1;
    }

    // ---- epilogue: D layout -> lane = col (lane&15), VGPR r -> row r or r+8
    const int col   = lane & 15;
    const int rbase = (lane >> 4) * 8;
#pragma unroll
    for (int i = 0; i < 4; ++i) {
#pragma unroll
        for (int j = 0; j < 2; ++j) {
            const int mbase = mt * 128 + wm * 64 + i * 16;
            const int nbase = nt * 128 + wn * 32 + j * 16 + col;
#pragma unroll
            for (int r = 0; r < 8; ++r) {
                const int mrow = mbase + rbase + r;      // row within expert
                float v = acc[i][j][r];
                if (FC1) {
                    v = gelu_exact(v);
                    Hout[((size_t)e * CAP + mrow) * 2048 + nbase] = f2bfu(v);
                } else {
                    const int tok = inv[e * CAP + mrow];
                    atomicAdd(&Out[(size_t)tok * HIDDEN + nbase], v);
                }
            }
        }
    }
}

// ---------------------------------------------------------------------
extern "C" void kernel_launch(void* const* d_in, const int* in_sizes, int n_in,
                              void* d_out, int out_size, void* d_ws, size_t ws_size,
                              hipStream_t stream) {
    const float* x    = (const float*)d_in[0];   // [NTOK, HIDDEN] fp32
    const float* w1   = (const float*)d_in[1];   // [NEXP, FFN, HIDDEN] fp32
    const float* w2   = (const float*)d_in[2];   // [NEXP, HIDDEN, FFN] fp32
    const int*   sidx = (const int*)d_in[3];     // [NTOK, TOPK]
    float*       out  = (float*)d_out;           // [NTOK, HIDDEN] fp32

    // workspace: x_scat bf16 | h bf16 | w1 bf16 | w2 bf16 | inv int  (~256 MB)
    unsigned short* xs  = (unsigned short*)d_ws;
    unsigned short* hb  = xs  + (size_t)MROWS * HIDDEN;
    unsigned short* w1b = hb  + (size_t)MROWS * FFN;
    unsigned short* w2b = w1b + (size_t)NEXP * FFN * HIDDEN;
    int*            inv = (int*)(w2b + (size_t)NEXP * HIDDEN * FFN);

    // fc2 accumulates -> out must start at zero every call
    hipMemsetAsync(out, 0, (size_t)out_size * sizeof(float), stream);

    // one-time weight downconvert (removes cvt VALU from GEMM hot loop)
    const long long nW = (long long)NEXP * FFN * HIDDEN;   // 33.5M each
    const int cvtBlocks = (int)(nW / (16 * 256));
    f32_to_bf16_kernel<<<cvtBlocks, 256, 0, stream>>>(w1, w1b, nW);
    f32_to_bf16_kernel<<<cvtBlocks, 256, 0, stream>>>(w2, w2b, nW);

    moe_scatter_bf16<<<MROWS, 256, 0, stream>>>(x, sidx, xs, inv);

    dim3 grid(CAP / 128, 2048 / 128, NEXP);      // 16 x 16 x 8
    moe_wmma_gemm<true ><<<grid, 256, 0, stream>>>(xs, w1b, hb, nullptr, nullptr);
    moe_wmma_gemm<false><<<grid, 256, 0, stream>>>(hb, w2b, nullptr, inv, out);
}
